// TransformerBlock_65481071399581
// MI455X (gfx1250) — compile-verified, matrix operand order checked by reading, not run on hardware
//
#include <hip/hip_runtime.h>
#include <stdint.h>

typedef __attribute__((ext_vector_type(16))) __bf16 v16bf;
typedef __attribute__((ext_vector_type(8)))  float  v8f;
typedef __attribute__((ext_vector_type(4)))  int    v4i;

union FragU { uint4 q[2]; v16bf v; };

#if defined(__has_builtin)
#if __has_builtin(__builtin_amdgcn_global_load_async_to_lds_b128)
#define HAVE_ASYNC_LDS 1
#endif
#endif

__device__ __forceinline__ unsigned short f2bf(float f) {
  union { __bf16 b; unsigned short u; } v;
  v.b = (__bf16)f;
  return v.u;
}

__device__ __forceinline__ v8f zero8() {
  v8f z;
#pragma unroll
  for (int i = 0; i < 8; ++i) z[i] = 0.f;
  return z;
}

// DPP-based reductions over each 16-lane row group (matches WMMA C layout:
// lanes 0-15 hold rows half*8+i for half=0, lanes 16-31 for half=1).
template <int CTRL>
__device__ __forceinline__ float dppf(float x) {
  union { float f; int i; } a, b;
  a.f = x;
  b.i = __builtin_amdgcn_mov_dpp(a.i, CTRL, 0xF, 0xF, false);
  return b.f;
}
__device__ __forceinline__ float row16_max(float x) {
  x = fmaxf(x, dppf<0xB1>(x));   // quad_perm(1,0,3,2)
  x = fmaxf(x, dppf<0x4E>(x));   // quad_perm(2,3,0,1)
  x = fmaxf(x, dppf<0x141>(x));  // row_half_mirror
  x = fmaxf(x, dppf<0x140>(x));  // row_mirror
  return x;
}
__device__ __forceinline__ float row16_sum(float x) {
  x += dppf<0xB1>(x);
  x += dppf<0x4E>(x);
  x += dppf<0x141>(x);
  x += dppf<0x140>(x);
  return x;
}

// 16-byte copy global -> LDS; async-to-LDS when the toolchain exposes it.
__device__ __forceinline__ void cp16(const unsigned short* g, unsigned short* l) {
#ifdef HAVE_ASYNC_LDS
  typedef __attribute__((address_space(1))) v4i* gp_t;
  typedef __attribute__((address_space(3))) v4i* lp_t;
  __builtin_amdgcn_global_load_async_to_lds_b128((gp_t)g, (lp_t)l, 0, 0);
#else
  *(uint4*)l = *(const uint4*)g;
#endif
}
__device__ __forceinline__ void wait_async() {
#ifdef HAVE_ASYNC_LDS
#if defined(__has_builtin) && __has_builtin(__builtin_amdgcn_s_wait_asynccnt)
  __builtin_amdgcn_s_wait_asynccnt(0);
#else
  asm volatile("s_wait_asynccnt 0x0" ::: "memory");
#endif
#endif
}

// Load a 16x32 bf16 WMMA fragment (A layout; B layout symmetric) from LDS.
// Lane (half = lane>>4, m = lane&15) holds K = {half*8..half*8+7} and
// {16+half*8..16+half*8+7}: two contiguous 16-byte chunks.
__device__ __forceinline__ v16bf load_frag(const unsigned short* base, int row,
                                           int ldm, int colbase, int half) {
  FragU f;
  const unsigned short* p = base + row * ldm + colbase + half * 8;
  f.q[0] = *(const uint4*)(p);
  f.q[1] = *(const uint4*)(p + 16);
  return f.v;
}

__device__ __forceinline__ v8f wmma_bf16(v16bf a, v16bf b, v8f c) {
  return __builtin_amdgcn_wmma_f32_16x16x32_bf16(false, a, false, b, (short)0,
                                                 c, false, false);
}

// ---------------------------------------------------------------------------
// fp32 -> bf16 conversion
// ---------------------------------------------------------------------------
__global__ void cvt_bf16_kernel(const float* __restrict__ in,
                                unsigned short* __restrict__ out, int n) {
  int i = blockIdx.x * blockDim.x + threadIdx.x;
  int stride = gridDim.x * blockDim.x;
  for (; i < n; i += stride) out[i] = f2bf(in[i]);
}

// ---------------------------------------------------------------------------
// GEMM: C[M,N] = A[M,K] * W[N,K]^T  (bf16 in, f32 accumulate)
// Block tile 128x128, K-step 32, 256 threads = 8 waves (4 M x 2 N),
// each wave owns a 32x64 tile = 2x4 WMMA accumulators.
// Double-buffered LDS: async loads of tile t+1 overlap WMMA on tile t.
// ---------------------------------------------------------------------------
template <bool OUT_BF16, bool HAS_BIAS, bool RELU>
__global__ __launch_bounds__(256) void gemm_wmma(
    const unsigned short* __restrict__ A, const unsigned short* __restrict__ W,
    const float* __restrict__ bias, void* __restrict__ Cout, int M, int N,
    int K) {
  __shared__ unsigned short sA[2][128 * 40];
  __shared__ unsigned short sW[2][128 * 40];
  const int tid = threadIdx.x;
  const int lane = tid & 31, w = tid >> 5;
  const int half = lane >> 4, l16 = lane & 15;
  const int mblk = blockIdx.y * 128, nblk = blockIdx.x * 128;
  const int mw = (w & 3) * 32, nw = (w >> 2) * 64;

  auto load_tiles = [&](int k0, int buf) {
#pragma unroll
    for (int it = 0; it < 2; ++it) {
      int idx = tid + it * 256;  // 0..511
      int r = idx >> 2, c = (idx & 3) * 8;
      cp16(&A[(size_t)(mblk + r) * K + k0 + c], &sA[buf][r * 40 + c]);
      cp16(&W[(size_t)(nblk + r) * K + k0 + c], &sW[buf][r * 40 + c]);
    }
  };

  v8f acc[2][4];
#pragma unroll
  for (int r = 0; r < 2; ++r)
#pragma unroll
    for (int c = 0; c < 4; ++c) acc[r][c] = zero8();

  load_tiles(0, 0);
  wait_async();
  __syncthreads();

  int cur = 0;
  for (int k0 = 0; k0 < K; k0 += 32) {
    int nxt = cur ^ 1;
    if (k0 + 32 < K) load_tiles(k0 + 32, nxt);  // overlap with compute below

    v16bf a[2], b[4];
#pragma unroll
    for (int r = 0; r < 2; ++r)
      a[r] = load_frag(sA[cur], mw + r * 16 + l16, 40, 0, half);
#pragma unroll
    for (int c = 0; c < 4; ++c)
      b[c] = load_frag(sW[cur], nw + c * 16 + l16, 40, 0, half);
#pragma unroll
    for (int r = 0; r < 2; ++r)
#pragma unroll
      for (int c = 0; c < 4; ++c) acc[r][c] = wmma_bf16(a[r], b[c], acc[r][c]);

    wait_async();
    __syncthreads();
    cur = nxt;
  }

  // Epilogue: C layout -> lane holds column (l16), VGPR slot i -> row half*8+i
#pragma unroll
  for (int r = 0; r < 2; ++r)
#pragma unroll
    for (int c = 0; c < 4; ++c) {
      int col = nblk + nw + c * 16 + l16;
      float bv = HAS_BIAS ? bias[col] : 0.f;
#pragma unroll
      for (int i = 0; i < 8; ++i) {
        int row = mblk + mw + r * 16 + half * 8 + i;
        float v = acc[r][c][i];
        if (HAS_BIAS) v += bv;
        if (RELU) v = fmaxf(v, 0.f);
        if (OUT_BF16)
          ((unsigned short*)Cout)[(size_t)row * N + col] = f2bf(v);
        else
          ((float*)Cout)[(size_t)row * N + col] = v;
      }
    }
}

// ---------------------------------------------------------------------------
// Flash attention, DK = 64, head-interleaved [B*S, E] bf16 q/k/v.
// Block: 128 queries for one (b,h); 8 waves, each wave owns 16 queries.
// Key step = 32; K/V staging double-buffered against compute.
// ---------------------------------------------------------------------------
__global__ __launch_bounds__(256) void flash_attn_kernel(
    const unsigned short* __restrict__ Q, const unsigned short* __restrict__ Kb,
    const unsigned short* __restrict__ Vb, unsigned short* __restrict__ ctx,
    int S, int E, int HH) {
  const int bh = blockIdx.y;
  const int b = bh / HH, h = bh % HH;
  const int qblk = blockIdx.x * 128;
  const int tid = threadIdx.x, lane = tid & 31, w = tid >> 5;
  const int half = lane >> 4, l16 = lane & 15;
  const size_t baseoff = (size_t)b * S * E + (size_t)h * 64;

  __shared__ unsigned short sQ[128 * 72];
  __shared__ unsigned short sK[2][32 * 72];
  __shared__ unsigned short sVt[2][64 * 40];  // transposed: [dim][key]
  __shared__ unsigned short sP[8 * 16 * 40];

#pragma unroll
  for (int it = 0; it < 4; ++it) {
    int idx = tid + it * 256;  // 0..1023 : 128 rows x 8 chunks
    int r = idx >> 3, d = (idx & 7) * 8;
    cp16(&Q[baseoff + (size_t)(qblk + r) * E + d], &sQ[r * 72 + d]);
  }

  auto load_kv = [&](int t, int buf) {
    int kbase = t * 32;
    int r = tid >> 3, d = (tid & 7) * 8;  // 32 keys x 64 dims
    cp16(&Kb[baseoff + (size_t)(kbase + r) * E + d], &sK[buf][r * 72 + d]);
    union { uint4 q; unsigned short s[8]; } vv;
    vv.q = *(const uint4*)&Vb[baseoff + (size_t)(kbase + r) * E + d];
#pragma unroll
    for (int ii = 0; ii < 8; ++ii) sVt[buf][(d + ii) * 40 + r] = vv.s[ii];
  };

  load_kv(0, 0);
  wait_async();
  __syncthreads();

  v16bf qf[2];
#pragma unroll
  for (int kc = 0; kc < 2; ++kc)
    qf[kc] = load_frag(sQ, w * 16 + l16, 72, kc * 32, half);

  v8f o[4];
#pragma unroll
  for (int f = 0; f < 4; ++f) o[f] = zero8();
  float mi[8], li[8];
#pragma unroll
  for (int i = 0; i < 8; ++i) { mi[i] = -1e30f; li[i] = 0.f; }
  const float scale = 0.125f;  // 1/sqrt(64)

  const int nsteps = S / 32;
  int cur = 0;
  for (int t = 0; t < nsteps; ++t) {
    int nxt = cur ^ 1;
    if (t + 1 < nsteps) load_kv(t + 1, nxt);  // overlap with compute below

    // scores: S[m, n] = sum_d Q[m,d] * K[n,d]
    v8f s0 = zero8(), s1 = zero8();
#pragma unroll
    for (int kc = 0; kc < 2; ++kc) {
      v16bf bk0 = load_frag(sK[cur], 0 + l16, 72, kc * 32, half);
      s0 = wmma_bf16(qf[kc], bk0, s0);
      v16bf bk1 = load_frag(sK[cur], 16 + l16, 72, kc * 32, half);
      s1 = wmma_bf16(qf[kc], bk1, s1);
    }

    // online softmax: slot i == row half*8+i, lane holds one column
    float p0[8], p1[8];
#pragma unroll
    for (int i = 0; i < 8; ++i) {
      float sv0 = s0[i] * scale, sv1 = s1[i] * scale;
      float rm = row16_max(fmaxf(sv0, sv1));
      float mnew = fmaxf(mi[i], rm);
      float corr = __expf(mi[i] - mnew);
      p0[i] = __expf(sv0 - mnew);
      p1[i] = __expf(sv1 - mnew);
      float rs = row16_sum(p0[i] + p1[i]);
      li[i] = li[i] * corr + rs;
      mi[i] = mnew;
#pragma unroll
      for (int f = 0; f < 4; ++f) o[f][i] *= corr;
    }

    // restage P (C layout -> A layout) through per-wave LDS slot
    unsigned short* sPw = &sP[w * 16 * 40];
#pragma unroll
    for (int i = 0; i < 8; ++i) {
      sPw[(half * 8 + i) * 40 + l16] = f2bf(p0[i]);
      sPw[(half * 8 + i) * 40 + 16 + l16] = f2bf(p1[i]);
    }
    __builtin_amdgcn_wave_barrier();
    asm volatile("s_wait_dscnt 0x0" ::: "memory");
    v16bf pa = load_frag(sPw, l16, 40, 0, half);
#pragma unroll
    for (int f = 0; f < 4; ++f) {
      v16bf bv = load_frag(sVt[cur], f * 16 + l16, 40, 0, half);
      o[f] = wmma_bf16(pa, bv, o[f]);
    }

    wait_async();
    __syncthreads();
    cur = nxt;
  }

#pragma unroll
  for (int i = 0; i < 8; ++i) {
    float inv = 1.f / li[i];
    int row = qblk + w * 16 + half * 8 + i;
#pragma unroll
    for (int f = 0; f < 4; ++f)
      ctx[baseoff + (size_t)row * E + f * 16 + l16] = f2bf(o[f][i] * inv);
  }
}

// ---------------------------------------------------------------------------
// y = LayerNorm(x + r) * g + beta ; optional bf16 copy of y. One block per row.
// ---------------------------------------------------------------------------
__global__ __launch_bounds__(256) void ln_residual_kernel(
    const float* __restrict__ X, const float* __restrict__ R,
    const float* __restrict__ g, const float* __restrict__ bta,
    float* __restrict__ Y, unsigned short* __restrict__ Yb, int E_) {
  const int row = blockIdx.x;
  const int tid = threadIdx.x;
  __shared__ float red[256];
  float v[4];
#pragma unroll
  for (int i = 0; i < 4; ++i) {
    int c = tid + i * 256;
    v[i] = X[(size_t)row * E_ + c] + R[(size_t)row * E_ + c];
  }
  red[tid] = v[0] + v[1] + v[2] + v[3];
  __syncthreads();
  for (int off = 128; off > 0; off >>= 1) {
    if (tid < off) red[tid] += red[tid + off];
    __syncthreads();
  }
  float mu = red[0] / (float)E_;
  __syncthreads();
  float vs = 0.f;
#pragma unroll
  for (int i = 0; i < 4; ++i) {
    float d = v[i] - mu;
    vs += d * d;
  }
  red[tid] = vs;
  __syncthreads();
  for (int off = 128; off > 0; off >>= 1) {
    if (tid < off) red[tid] += red[tid + off];
    __syncthreads();
  }
  float rstd = rsqrtf(red[0] / (float)E_ + 1e-5f);
#pragma unroll
  for (int i = 0; i < 4; ++i) {
    int c = tid + i * 256;
    float yv = (v[i] - mu) * rstd * g[c] + bta[c];
    Y[(size_t)row * E_ + c] = yv;
    if (Yb) Yb[(size_t)row * E_ + c] = f2bf(yv);
  }
}

// ---------------------------------------------------------------------------
extern "C" void kernel_launch(void* const* d_in, const int* in_sizes, int n_in,
                              void* d_out, int out_size, void* d_ws,
                              size_t ws_size, hipStream_t stream) {
  (void)in_sizes; (void)n_in; (void)out_size; (void)ws_size;
  const float* x  = (const float*)d_in[0];
  const float* wq = (const float*)d_in[1];
  const float* wk = (const float*)d_in[2];
  const float* wv = (const float*)d_in[3];
  const float* wo = (const float*)d_in[4];
  const float* w1 = (const float*)d_in[5];
  const float* b1 = (const float*)d_in[6];
  const float* w2 = (const float*)d_in[7];
  const float* b2 = (const float*)d_in[8];
  const float* g1 = (const float*)d_in[9];
  const float* be1 = (const float*)d_in[10];
  const float* g2 = (const float*)d_in[11];
  const float* be2 = (const float*)d_in[12];

  const int B = 4, S = 2048, E = 1024, H = 16, FFN = 4096;
  const int Mtok = B * S;  // 8192
  const size_t MB = 1ull << 20;

  char* ws = (char*)d_ws;
  unsigned short* xb  = (unsigned short*)(ws + 0);        // 16 MB
  unsigned short* wqb = (unsigned short*)(ws + 16 * MB);  // 2 MB
  unsigned short* wkb = (unsigned short*)(ws + 18 * MB);
  unsigned short* wvb = (unsigned short*)(ws + 20 * MB);
  unsigned short* wob = (unsigned short*)(ws + 22 * MB);
  unsigned short* w1b = (unsigned short*)(ws + 24 * MB);  // 8 MB
  unsigned short* w2b = (unsigned short*)(ws + 32 * MB);  // 8 MB
  unsigned short* qb  = (unsigned short*)(ws + 40 * MB);  // 16 MB
  unsigned short* kb  = (unsigned short*)(ws + 56 * MB);  // 16 MB
  unsigned short* vb  = (unsigned short*)(ws + 72 * MB);  // 16 MB
  unsigned short* ctxb = (unsigned short*)(ws + 88 * MB); // 16 MB
  unsigned short* hb  = (unsigned short*)(ws + 40 * MB);  // 64 MB, reuses q/k/v/ctx
  float* ao  = (float*)(ws + 104 * MB);                   // 32 MB
  float* ffn = (float*)(ws + 104 * MB);                   // reuses ao
  float* x1  = (float*)(ws + 136 * MB);                   // 32 MB
  unsigned short* x1b = (unsigned short*)(ws + 168 * MB); // 16 MB

  cvt_bf16_kernel<<<2048, 256, 0, stream>>>(x, xb, Mtok * E);
  cvt_bf16_kernel<<<1024, 256, 0, stream>>>(wq, wqb, E * E);
  cvt_bf16_kernel<<<1024, 256, 0, stream>>>(wk, wkb, E * E);
  cvt_bf16_kernel<<<1024, 256, 0, stream>>>(wv, wvb, E * E);
  cvt_bf16_kernel<<<1024, 256, 0, stream>>>(wo, wob, E * E);
  cvt_bf16_kernel<<<2048, 256, 0, stream>>>(w1, w1b, FFN * E);
  cvt_bf16_kernel<<<2048, 256, 0, stream>>>(w2, w2b, E * FFN);

  dim3 gEE(E / 128, Mtok / 128);
  gemm_wmma<true, false, false><<<gEE, 256, 0, stream>>>(xb, wqb, nullptr, qb, Mtok, E, E);
  gemm_wmma<true, false, false><<<gEE, 256, 0, stream>>>(xb, wkb, nullptr, kb, Mtok, E, E);
  gemm_wmma<true, false, false><<<gEE, 256, 0, stream>>>(xb, wvb, nullptr, vb, Mtok, E, E);

  flash_attn_kernel<<<dim3(S / 128, B * H), 256, 0, stream>>>(qb, kb, vb, ctxb, S, E, H);

  gemm_wmma<false, false, false><<<gEE, 256, 0, stream>>>(ctxb, wob, nullptr, ao, Mtok, E, E);

  ln_residual_kernel<<<Mtok, 256, 0, stream>>>(x, ao, g1, be1, x1, x1b, E);

  gemm_wmma<true, true, true><<<dim3(FFN / 128, Mtok / 128), 256, 0, stream>>>(
      x1b, w1b, b1, hb, Mtok, FFN, E);

  gemm_wmma<false, true, false><<<gEE, 256, 0, stream>>>(hb, w2b, b2, ffn, Mtok, E, FFN);

  ln_residual_kernel<<<Mtok, 256, 0, stream>>>(x1, ffn, g2, be2, (float*)d_out, nullptr, E);
}
